// TransformerEncoder_18494129176577
// MI455X (gfx1250) — matrix operand order repaired and checked by reading, not compiled, not run on hardware
//
#include <hip/hip_runtime.h>
#include <hip/hip_bf16.h>

#define NB   2        // batch
#define SQ   2048     // sequence
#define DM   768      // model dim
#define NHD  12       // heads
#define FF_  3072     // ffn dim
#define NL   2        // layers
#define DH   64       // head dim
#define EPSV 1e-5f

typedef __attribute__((ext_vector_type(16))) __bf16 v16bf;
typedef __attribute__((ext_vector_type(8)))  __bf16 v8bf;
typedef __attribute__((ext_vector_type(8)))  float  v8f;

__device__ __forceinline__ __bf16 to_bf(float f) { return (__bf16)f; }

// ---------------------------------------------------------------------------
// Batched WMMA GEMM:  C = act( scale * (A @ Bt^T) + bias )
//  A : f32  [M,K] row-major (lda)   (converted to bf16 in-register)
//  Bt: bf16 [N,K] row-major (ldb)   (pre-transposed, K contiguous per row)
//  Per-z offsets: off = (z/Hdiv)*xB + (z%Hdiv)*xH   (head-batched attention)
//  Block = 128 threads = 4 waves; each wave -> 32x32 tile; block tile 64x64.
// ---------------------------------------------------------------------------
template <bool RELU>
__global__ __launch_bounds__(128) void gemm_wmma(
    const float* __restrict__ A, const __bf16* __restrict__ Bm,
    const float* __restrict__ bias, float* __restrict__ C,
    int M, int N, int K, int lda, int ldb, int ldc, float scale, int Hdiv,
    long long aB, long long aH, long long bB, long long bH,
    long long cB, long long cH)
{
    const int z  = blockIdx.z;
    const int zb = z / Hdiv, zh = z % Hdiv;
    const float*  Ab = A  + (long long)zb * aB + (long long)zh * aH;
    const __bf16* Bb = Bm + (long long)zb * bB + (long long)zh * bH;
    float*        Cb = C  + (long long)zb * cB + (long long)zh * cH;

    const int lane = threadIdx.x & 31;
    const int wave = threadIdx.x >> 5;
    const int wm = wave >> 1, wn = wave & 1;
    const int m0 = blockIdx.y * 64 + wm * 32;
    const int n0 = blockIdx.x * 64 + wn * 32;
    const int lm = lane & 15;   // row/col index within 16
    const int lh = lane >> 4;   // lane half (0/1)

    v8f acc[2][2] = {};

    for (int k0 = 0; k0 < K; k0 += 32) {
        v16bf afrag[2], bfrag[2];
        // A fragment (16x32 bf16): lane lm holds row M=lm.
        // elems 0..7  -> K = k0 + lh*8      + e
        // elems 8..15 -> K = k0 + 16 + lh*8 + e   (ISA 7.12.2, 16-bit A 16x32)
        #pragma unroll
        for (int t = 0; t < 2; ++t) {
            const float* ap = Ab + (long long)(m0 + t * 16 + lm) * lda + k0 + lh * 8;
            #pragma unroll
            for (int e = 0; e < 8; ++e) afrag[t][e]     = to_bf(ap[e]);
            #pragma unroll
            for (int e = 0; e < 8; ++e) afrag[t][8 + e] = to_bf(ap[16 + e]);
            __builtin_prefetch(ap + 32, 0, 0);   // next k-tile (global_prefetch_b8)
        }
        // B fragment (32x16 bf16): lane lm holds col N=lm; same K order as A
        #pragma unroll
        for (int t = 0; t < 2; ++t) {
            const __bf16* bp = Bb + (long long)(n0 + t * 16 + lm) * ldb + k0 + lh * 8;
            const v8bf lo = *(const v8bf*)bp;
            const v8bf hi = *(const v8bf*)(bp + 16);
            #pragma unroll
            for (int e = 0; e < 8; ++e) { bfrag[t][e] = lo[e]; bfrag[t][8 + e] = hi[e]; }
        }
        #pragma unroll
        for (int tm = 0; tm < 2; ++tm)
            #pragma unroll
            for (int tn = 0; tn < 2; ++tn)
                acc[tm][tn] = __builtin_amdgcn_wmma_f32_16x16x32_bf16(
                    false, afrag[tm], false, bfrag[tn],
                    (short)0, acc[tm][tn], false, false);
    }

    // Epilogue. D layout: element r -> row m = r + 8*lh, col n = lm.
    #pragma unroll
    for (int tm = 0; tm < 2; ++tm) {
        #pragma unroll
        for (int tn = 0; tn < 2; ++tn) {
            const int cn = n0 + tn * 16 + lm;
            const float bv = bias ? bias[cn] : 0.0f;
            #pragma unroll
            for (int r = 0; r < 8; ++r) {
                const int cm = m0 + tm * 16 + r + lh * 8;
                float v = acc[tm][tn][r] * scale + bv;
                if (RELU) v = fmaxf(v, 0.0f);
                Cb[(long long)cm * ldc + cn] = v;
            }
        }
    }
}

// ---------------------------------------------------------------------------
// f32 -> bf16 bulk conversion (grid-stride)
// ---------------------------------------------------------------------------
__global__ __launch_bounds__(256) void f32_to_bf16(
    const float* __restrict__ s, __bf16* __restrict__ d, long long n)
{
    const long long stride = (long long)gridDim.x * 256;
    for (long long i = (long long)blockIdx.x * 256 + threadIdx.x; i < n; i += stride)
        d[i] = (__bf16)s[i];
}

// ---------------------------------------------------------------------------
// Tiled transpose + f32 -> bf16:  dst[c][r] = src[r][c]
//  src: f32 [rows, cols], dst: bf16 [cols, rows]; rows, cols % 32 == 0.
//  Batched via blockIdx.z.  32x33 LDS tile (no bank conflicts).
// ---------------------------------------------------------------------------
__global__ __launch_bounds__(256) void transpose_cvt(
    const float* __restrict__ src, __bf16* __restrict__ dst,
    int rows, int cols, long long sBatch, long long dBatch)
{
    __shared__ float tile[32][33];
    const float* s = src + (long long)blockIdx.z * sBatch;
    __bf16*      d = dst + (long long)blockIdx.z * dBatch;
    const int c0 = blockIdx.x * 32;
    const int r0 = blockIdx.y * 32;
    const int tx = threadIdx.x & 31;
    const int ty = threadIdx.x >> 5;   // 0..7
    #pragma unroll
    for (int i = 0; i < 4; ++i)
        tile[ty + i * 8][tx] = s[(long long)(r0 + ty + i * 8) * cols + c0 + tx];
    __syncthreads();
    #pragma unroll
    for (int i = 0; i < 4; ++i)
        d[(long long)(c0 + ty + i * 8) * rows + r0 + tx] = (__bf16)tile[tx][ty + i * 8];
}

// ---------------------------------------------------------------------------
// Embedding + sinusoidal positional encoding: h[row, :] = emb[x[row], :] + pe
// ---------------------------------------------------------------------------
__global__ __launch_bounds__(256) void embed_pe(
    const int* __restrict__ x, const float* __restrict__ emb, float* __restrict__ h)
{
    const long long row = blockIdx.x;             // b*SQ + s
    const int s = (int)(row % SQ);
    const int tok = x[row];
    const float* ep = emb + (long long)tok * DM;
    float* hp = h + row * DM;
    for (int j = threadIdx.x; j < DM; j += 256) {
        const int i2 = j & ~1;
        const float freq = expf(-0.011992631f * (float)i2);  // ln(10000)/768
        const float ang = (float)s * freq;
        const float pe = (j & 1) ? cosf(ang) : sinf(ang);
        hp[j] = ep[j] + pe;
    }
}

// ---------------------------------------------------------------------------
// In-place row softmax (row length = SQ, 256 threads, 8 elems/thread)
// ---------------------------------------------------------------------------
__global__ __launch_bounds__(256) void softmax_inplace(float* __restrict__ data)
{
    __shared__ float sh[256];
    const long long row = blockIdx.x;
    float* p = data + row * (long long)SQ;
    const int t = threadIdx.x;
    float vals[8];
    float mx = -3.0e38f;
    #pragma unroll
    for (int i = 0; i < 8; ++i) { vals[i] = p[t + i * 256]; mx = fmaxf(mx, vals[i]); }
    sh[t] = mx; __syncthreads();
    for (int k = 128; k > 0; k >>= 1) { if (t < k) sh[t] = fmaxf(sh[t], sh[t + k]); __syncthreads(); }
    mx = sh[0]; __syncthreads();
    float sum = 0.0f;
    #pragma unroll
    for (int i = 0; i < 8; ++i) { vals[i] = __expf(vals[i] - mx); sum += vals[i]; }
    sh[t] = sum; __syncthreads();
    for (int k = 128; k > 0; k >>= 1) { if (t < k) sh[t] += sh[t + k]; __syncthreads(); }
    const float inv = 1.0f / sh[0];
    #pragma unroll
    for (int i = 0; i < 8; ++i) p[t + i * 256] = vals[i] * inv;
}

// ---------------------------------------------------------------------------
// h = LayerNorm(h + res) * g + b   (row length DM = 768, 3 elems/thread)
// ---------------------------------------------------------------------------
__global__ __launch_bounds__(256) void add_ln(
    float* __restrict__ h, const float* __restrict__ res,
    const float* __restrict__ g, const float* __restrict__ b)
{
    __shared__ float sh[256];
    const long long row = blockIdx.x;
    float* hp = h + row * DM;
    const float* rp = res + row * DM;
    const int t = threadIdx.x;
    float v[3];
    #pragma unroll
    for (int i = 0; i < 3; ++i) v[i] = hp[t + i * 256] + rp[t + i * 256];
    sh[t] = v[0] + v[1] + v[2]; __syncthreads();
    for (int k = 128; k > 0; k >>= 1) { if (t < k) sh[t] += sh[t + k]; __syncthreads(); }
    const float mu = sh[0] / (float)DM; __syncthreads();
    float ss = 0.0f;
    #pragma unroll
    for (int i = 0; i < 3; ++i) { const float d = v[i] - mu; ss += d * d; }
    sh[t] = ss; __syncthreads();
    for (int k = 128; k > 0; k >>= 1) { if (t < k) sh[t] += sh[t + k]; __syncthreads(); }
    const float rstd = rsqrtf(sh[0] / (float)DM + EPSV);
    #pragma unroll
    for (int i = 0; i < 3; ++i) {
        const int j = t + i * 256;
        hp[j] = (v[i] - mu) * rstd * g[j] + b[j];
    }
}

__global__ void copy_f32(const float* __restrict__ src, float* __restrict__ dst, long long n)
{
    const long long i = (long long)blockIdx.x * blockDim.x + threadIdx.x;
    if (i < n) dst[i] = src[i];
}

// ---------------------------------------------------------------------------
static inline void gemm(hipStream_t st, const float* A, const __bf16* Bt, const float* bias,
                        float* C, int M, int N, int K, int lda, int ldb, int ldc,
                        float scale, bool relu, int batch, int Hdiv,
                        long long aB, long long aH, long long bB, long long bH,
                        long long cB, long long cH)
{
    dim3 grid(N / 64, M / 64, batch), blk(128);
    if (relu) gemm_wmma<true ><<<grid, blk, 0, st>>>(A, Bt, bias, C, M, N, K, lda, ldb, ldc, scale, Hdiv, aB, aH, bB, bH, cB, cH);
    else      gemm_wmma<false><<<grid, blk, 0, st>>>(A, Bt, bias, C, M, N, K, lda, ldb, ldc, scale, Hdiv, aB, aH, bB, bH, cB, cH);
}

static inline void cvt(hipStream_t st, const float* s, __bf16* d, long long n)
{
    f32_to_bf16<<<(int)((n + 255) / 256), 256, 0, st>>>(s, d, n);
}

static inline void tcvt(hipStream_t st, const float* s, __bf16* d, int rows, int cols,
                        int batch, long long sB, long long dB)
{
    dim3 grid(cols / 32, rows / 32, batch), blk(256);
    transpose_cvt<<<grid, blk, 0, st>>>(s, d, rows, cols, sB, dB);
}

extern "C" void kernel_launch(void* const* d_in, const int* in_sizes, int n_in,
                              void* d_out, int out_size, void* d_ws, size_t ws_size,
                              hipStream_t stream)
{
    const int*   x    = (const int*)  d_in[0];
    const float* emb  = (const float*)d_in[1];
    const float* Wq   = (const float*)d_in[2];
    const float* bq   = (const float*)d_in[3];
    const float* Wk   = (const float*)d_in[4];
    const float* bk   = (const float*)d_in[5];
    const float* Wv   = (const float*)d_in[6];
    const float* bv   = (const float*)d_in[7];
    const float* Wo   = (const float*)d_in[8];
    const float* bo   = (const float*)d_in[9];
    const float* ln1g = (const float*)d_in[10];
    const float* ln1b = (const float*)d_in[11];
    const float* ln2g = (const float*)d_in[12];
    const float* ln2b = (const float*)d_in[13];
    const float* W1   = (const float*)d_in[14];
    const float* b1   = (const float*)d_in[15];
    const float* W2   = (const float*)d_in[16];
    const float* b2   = (const float*)d_in[17];

    float* out = (float*)d_out;
    const long long NHEL = (long long)NB * SQ * DM;   // 3,145,728 (final h)
    const long long SS   = (long long)SQ * SQ;        // 4,194,304
    const long long BHSS = (long long)NB * NHD * SS;  // 100,663,296 per layer
    const long long DD   = (long long)DM * DM;
    const long long DF   = (long long)DM * FF_;
    float* maps = out + NHEL;

    // Workspace layout: f32 region then bf16 region
    float* ws = (float*)d_ws;
    float* h  = ws;
    float* q  = h  + NHEL;
    float* kk = q  + NHEL;
    float* vv = kk + NHEL;
    float* o  = vv + NHEL;
    float* t2 = o  + NHEL;
    float* ff = t2 + NHEL;                 // NB*SQ*FF_ floats
    __bf16* bfbase = (__bf16*)(ff + (long long)NB * SQ * FF_);
    __bf16* Wq_t = bfbase;                 // NL*[DM,DM] transposed
    __bf16* Wk_t = Wq_t + NL * DD;
    __bf16* Wv_t = Wk_t + NL * DD;
    __bf16* Wo_t = Wv_t + NL * DD;
    __bf16* W1_t = Wo_t + NL * DD;         // NL*[FF_,DM]  (transpose of [DM,FF_])
    __bf16* W2_t = W1_t + NL * DF;         // NL*[DM,FF_]  (transpose of [FF_,DM])
    __bf16* k_bf = W2_t + NL * DF;         // [NB*SQ, DM]  (row-major, used as Bt)
    __bf16* v_t  = k_bf + NHEL;            // NB*[DM, SQ]  (transpose per batch)

    const int BS = NB * SQ;  // 4096 rows
    const long long SD = (long long)SQ * DM;

    // One-time transposed weight conversions (amortized over ~128x B-tile reuse)
    tcvt(stream, Wq, Wq_t, DM, DM, NL, DD, DD);
    tcvt(stream, Wk, Wk_t, DM, DM, NL, DD, DD);
    tcvt(stream, Wv, Wv_t, DM, DM, NL, DD, DD);
    tcvt(stream, Wo, Wo_t, DM, DM, NL, DD, DD);
    tcvt(stream, W1, W1_t, DM, FF_, NL, DF, DF);
    tcvt(stream, W2, W2_t, FF_, DM, NL, DF, DF);

    embed_pe<<<BS, 256, 0, stream>>>(x, emb, h);

    for (int l = 0; l < NL; ++l) {
        // Q/K/V projections: [4096,768] = h @ W + b   (Bt = Wt[N=DM, K=DM])
        gemm(stream, h, Wq_t + l * DD, bq + (long long)l * DM, q,  BS, DM, DM, DM, DM, DM, 1.0f, false, 1, 1, 0,0,0,0,0,0);
        gemm(stream, h, Wk_t + l * DD, bk + (long long)l * DM, kk, BS, DM, DM, DM, DM, DM, 1.0f, false, 1, 1, 0,0,0,0,0,0);
        gemm(stream, h, Wv_t + l * DD, bv + (long long)l * DM, vv, BS, DM, DM, DM, DM, DM, 1.0f, false, 1, 1, 0,0,0,0,0,0);

        // bf16 K (already Bt layout for QK^T); transposed bf16 V per batch
        cvt(stream, kk, k_bf, NHEL);
        tcvt(stream, vv, v_t, SQ, DM, NB, SD, SD);

        // scores = Q K^T / 8 -> d_out maps[l]; z = b*H + head
        float* mapsl = maps + (long long)l * BHSS;
        gemm(stream, q, k_bf, nullptr, mapsl, SQ, SQ, DH, DM, DM, SQ, 0.125f,
             false, NB * NHD, NHD, SD, DH, SD, DH, (long long)NHD * SS, SS);

        softmax_inplace<<<NB * NHD * SQ, 256, 0, stream>>>(mapsl);

        // o = attn @ V  (Bt = v_t[b] + head*DH*SQ, ldb = SQ); writes [B,S,D] interleave
        gemm(stream, mapsl, v_t, nullptr, o, SQ, DH, SQ, SQ, SQ, DM, 1.0f,
             false, NB * NHD, NHD, (long long)NHD * SS, SS,
             (long long)DM * SQ, (long long)DH * SQ, SD, DH);

        // attn_out = o @ Wo + bo ; h = LN(h + attn_out)
        gemm(stream, o, Wo_t + l * DD, bo + (long long)l * DM, t2, BS, DM, DM, DM, DM, DM, 1.0f, false, 1, 1, 0,0,0,0,0,0);
        add_ln<<<BS, 256, 0, stream>>>(h, t2, ln1g + (long long)l * DM, ln1b + (long long)l * DM);

        // FFN: ff = relu(h @ W1 + b1); t2 = ff @ W2 + b2; h = LN(h + t2)
        gemm(stream, h,  W1_t + l * DF, b1 + (long long)l * FF_, ff, BS, FF_, DM, DM, DM, FF_, 1.0f, true,  1, 1, 0,0,0,0,0,0);
        gemm(stream, ff, W2_t + l * DF, b2 + (long long)l * DM,  t2, BS, DM, FF_, FF_, FF_, DM, 1.0f, false, 1, 1, 0,0,0,0,0,0);
        add_ln<<<BS, 256, 0, stream>>>(h, t2, ln2g + (long long)l * DM, ln2b + (long long)l * DM);
    }

    copy_f32<<<(int)((NHEL + 255) / 256), 256, 0, stream>>>(h, out, NHEL);
}
